// SimpleSelfAttentionMask_52785148068063
// MI455X (gfx1250) — matrix operand order repaired and checked by reading, not compiled, not run on hardware
//
#include <hip/hip_runtime.h>
#include <hip/hip_bf16.h>

// ---------------- CDNA5 WMMA plumbing ----------------
typedef __attribute__((ext_vector_type(16))) __bf16 v16bf;
typedef __attribute__((ext_vector_type(8)))  float  v8f;

union FragAB { v16bf v; uint4 q[2]; __bf16 h[16]; };
union FragC  { v8f   v; float f[8]; };

__device__ __forceinline__ v16bf load_frag(const __bf16* p0, const __bf16* p1) {
  FragAB f;
  f.q[0] = *(const uint4*)p0;   // 8 contiguous bf16 (16B)
  f.q[1] = *(const uint4*)p1;   // 8 contiguous bf16 (16B)
  return f.v;
}

__device__ __forceinline__ v8f wmma_bf16(v16bf a, v16bf b, v8f c) {
  // D(16x16,f32) = A(16x32,bf16) x B(32x16,bf16) + C
  return __builtin_amdgcn_wmma_f32_16x16x32_bf16(false, a, false, b, (short)0, c,
                                                 false, false);
}

// ---- optional CDNA5 async global->LDS path (ASYNCcnt-tracked DMA) ----
#if defined(__AMDGCN__) && defined(__has_builtin)
#if __has_builtin(__builtin_amdgcn_global_load_async_to_lds_b128) && \
    __has_builtin(__builtin_amdgcn_s_wait_asynccnt)
#define USE_ASYNC_LDS 1
#endif
#endif
#ifndef USE_ASYNC_LDS
#define USE_ASYNC_LDS 0
#endif

#if USE_ASYNC_LDS
typedef int v4i_raw __attribute__((vector_size(16)));
typedef __attribute__((address_space(1))) v4i_raw* glb_v4i;
typedef __attribute__((address_space(3))) v4i_raw* lds_v4i;
__device__ __forceinline__ void async_copy16(const __bf16* g, __bf16* l) {
  __builtin_amdgcn_global_load_async_to_lds_b128((glb_v4i)g, (lds_v4i)l, 0, 0);
}
#endif

#define S_LEN 2048
#define F_DIM 1024
#define H_NUM 16
#define D_DIM 64

// ---------------- LayerNorm (optionally emits bf16 copy) ----------------
__global__ __launch_bounds__(256)
void ln_kernel(const float* __restrict__ x, const float* __restrict__ gamma,
               const float* __restrict__ beta, float* __restrict__ out32,
               __bf16* __restrict__ out16) {
  const int F = F_DIM;
  int row = blockIdx.x;
  const float* xr = x + (size_t)row * F;
  __shared__ float s1[256], s2[256];
  float a = 0.f, b2 = 0.f;
  for (int i = threadIdx.x; i < F; i += 256) { float v = xr[i]; a += v; b2 += v * v; }
  s1[threadIdx.x] = a; s2[threadIdx.x] = b2;
  __syncthreads();
  for (int st = 128; st > 0; st >>= 1) {
    if (threadIdx.x < st) { s1[threadIdx.x] += s1[threadIdx.x + st];
                            s2[threadIdx.x] += s2[threadIdx.x + st]; }
    __syncthreads();
  }
  float mu = s1[0] * (1.0f / F);
  float var = s2[0] * (1.0f / F) - mu * mu;
  float rstd = rsqrtf(var + 1e-6f);
  for (int i = threadIdx.x; i < F; i += 256) {
    float v = (xr[i] - mu) * rstd * gamma[i] + beta[i];
    out32[(size_t)row * F + i] = v;
    if (out16) out16[(size_t)row * F + i] = (__bf16)v;
  }
}

// ---------------- fp32 W[K][N] -> bf16 W^T[N][K] (one-time) ----------------
__global__ __launch_bounds__(256)
void transpose_w_bf16(const float* __restrict__ src, __bf16* __restrict__ dst) {
  __shared__ float tile[32][33];
  int bx = blockIdx.x * 32, by = blockIdx.y * 32;
  int tx = threadIdx.x & 31, ty = threadIdx.x >> 5;    // 32x8 threads
  for (int r = ty; r < 32; r += 8)
    tile[r][tx] = src[(size_t)(by + r) * F_DIM + bx + tx];
  __syncthreads();
  for (int r = ty; r < 32; r += 8)
    dst[(size_t)(bx + r) * F_DIM + by + tx] = (__bf16)tile[tx][r];
}

// ---------------- bf16 V[BH][S][D] -> V^T[BH][D][S] (one-time) ----------------
__global__ __launch_bounds__(256)
void transpose_v_bf16(const __bf16* __restrict__ src, __bf16* __restrict__ dst) {
  __shared__ __bf16 tile[32][33];
  int bh = blockIdx.z;
  int s0 = blockIdx.x * 32, d0 = blockIdx.y * 32;
  int tx = threadIdx.x & 31, ty = threadIdx.x >> 5;
  const __bf16* sb = src + (size_t)bh * S_LEN * D_DIM;
  __bf16* db = dst + (size_t)bh * D_DIM * S_LEN;
  for (int r = ty; r < 32; r += 8)
    tile[r][tx] = sb[(size_t)(s0 + r) * D_DIM + d0 + tx];
  __syncthreads();
  for (int r = ty; r < 32; r += 8)
    db[(size_t)(d0 + r) * S_LEN + s0 + tx] = tile[tx][r];
}

// ---------------- zero fill (attn buffer) ----------------
__global__ void zero_f4(float4* __restrict__ p, long n4) {
  long i = (long)blockIdx.x * blockDim.x + threadIdx.x;
  long stride = (long)gridDim.x * blockDim.x;
  float4 z = make_float4(0.f, 0.f, 0.f, 0.f);
  for (; i < n4; i += stride) p[i] = z;
}

// ---------------- WMMA GEMM: C[M,N] = A[M,K] * W[K,N] + epilogue ----------------
// Wt is W^T[N][K] bf16 (pre-transposed: B-fragment layout is row-major slices).
// Double-buffered LDS staging: tile t+1 fetched (async DMA if available) while
// tile t feeds the WMMAs; single barrier per k-step.
// mode 0: Q proj  -> bf16 [B,H,S,D] (+bias)
// mode 1: K proj  -> bf16 [B,H,S,D] (+bias +pos_emb[s,d])
// mode 2: V proj  -> bf16 [B,H,S,D] (+bias)
// mode 3: O proj  -> fp32 [M,F]     (+bias +resid)
__global__ __launch_bounds__(256)
void gemm_wmma(const __bf16* __restrict__ A, const __bf16* __restrict__ Wt,
               const float* __restrict__ bias, const float* __restrict__ posemb,
               const float* __restrict__ resid, __bf16* __restrict__ outb,
               float* __restrict__ outf, int mode) {
  const int F = F_DIM, Kdim = F_DIM, S = S_LEN, D = D_DIM, H = H_NUM;
  const int BK = 32, NKB = F_DIM / BK;
  int n0 = blockIdx.x * 64, m0 = blockIdx.y * 128;
  __shared__ __align__(16) __bf16 As[2][128][32];    // A tiles, row-major
  __shared__ __align__(16) __bf16 Bts[2][64][32];    // B^T tiles (row n, col k)
  int t = threadIdx.x, lane = t & 31, wave = t >> 5;
  int lane16 = lane & 15, hi = lane >> 4;

  v8f acc[4];
  v8f vz = {0.f, 0.f, 0.f, 0.f, 0.f, 0.f, 0.f, 0.f};
#pragma unroll
  for (int j = 0; j < 4; j++) acc[j] = vz;

  // per-thread source pointers for cooperative tile copies
  int arow = t >> 1, ahalf = t & 1;                       // 128 rows x 2 halves
  const __bf16* asrc = A + (size_t)(m0 + arow) * Kdim + ahalf * 16;
  int brow = t >> 2, bpart = t & 3;                       // 64 rows x 4 parts
  const __bf16* bsrc = Wt + (size_t)(n0 + brow) * Kdim + bpart * 8;

  // stage tile 0 into buffer 0
#if USE_ASYNC_LDS
  async_copy16(asrc,     &As[0][arow][ahalf * 16]);
  async_copy16(asrc + 8, &As[0][arow][ahalf * 16 + 8]);
  async_copy16(bsrc,     &Bts[0][brow][bpart * 8]);
  __builtin_amdgcn_s_wait_asynccnt(0);
#else
  *(uint4*)&As[0][arow][ahalf * 16]     = *(const uint4*)(asrc);
  *(uint4*)&As[0][arow][ahalf * 16 + 8] = *(const uint4*)(asrc + 8);
  *(uint4*)&Bts[0][brow][bpart * 8]     = *(const uint4*)(bsrc);
#endif
  __syncthreads();

  for (int it = 0; it < NKB; ++it) {
    int cur = it & 1, nxt = cur ^ 1;
    int kb = it * BK;
    const __bf16* an = asrc + kb + BK;
    const __bf16* bn = bsrc + kb + BK;

#if USE_ASYNC_LDS
    if (it + 1 < NKB) {           // DMA next tile while computing this one
      async_copy16(an,     &As[nxt][arow][ahalf * 16]);
      async_copy16(an + 8, &As[nxt][arow][ahalf * 16 + 8]);
      async_copy16(bn,     &Bts[nxt][brow][bpart * 8]);
    }
#else
    uint4 ua0, ua1, ub;
    if (it + 1 < NKB) {           // issue global loads early (hide under WMMA)
      __builtin_prefetch(an + BK, 0, 1);   // tile t+2 -> cache
      __builtin_prefetch(bn + BK, 0, 1);
      ua0 = *(const uint4*)(an);
      ua1 = *(const uint4*)(an + 8);
      ub  = *(const uint4*)(bn);
    }
#endif

    // compute on current buffer
    v16bf a = load_frag(&As[cur][wave * 16 + lane16][hi * 16],
                        &As[cur][wave * 16 + lane16][hi * 16 + 8]);
    v16bf b0 = load_frag(&Bts[cur][ 0 + lane16][hi * 16], &Bts[cur][ 0 + lane16][hi * 16 + 8]);
    v16bf b1 = load_frag(&Bts[cur][16 + lane16][hi * 16], &Bts[cur][16 + lane16][hi * 16 + 8]);
    v16bf b2 = load_frag(&Bts[cur][32 + lane16][hi * 16], &Bts[cur][32 + lane16][hi * 16 + 8]);
    v16bf b3 = load_frag(&Bts[cur][48 + lane16][hi * 16], &Bts[cur][48 + lane16][hi * 16 + 8]);
    acc[0] = wmma_bf16(a, b0, acc[0]);
    acc[1] = wmma_bf16(a, b1, acc[1]);
    acc[2] = wmma_bf16(a, b2, acc[2]);
    acc[3] = wmma_bf16(a, b3, acc[3]);

#if USE_ASYNC_LDS
    if (it + 1 < NKB) __builtin_amdgcn_s_wait_asynccnt(0);
#else
    if (it + 1 < NKB) {
      *(uint4*)&As[nxt][arow][ahalf * 16]     = ua0;
      *(uint4*)&As[nxt][arow][ahalf * 16 + 8] = ua1;
      *(uint4*)&Bts[nxt][brow][bpart * 8]     = ub;
    }
#endif
    __syncthreads();
  }

#pragma unroll
  for (int j = 0; j < 4; j++) {
    FragC c; c.v = acc[j];
#pragma unroll
    for (int r = 0; r < 8; r++) {
      int gm = m0 + wave * 16 + hi * 8 + r;      // token row
      int gn = n0 + j * 16 + lane16;             // feature col
      float val = c.f[r] + bias[gn];
      if (mode == 3) {
        val += resid[(size_t)gm * F + gn];
        outf[(size_t)gm * F + gn] = val;
      } else {
        int d = gn & (D - 1), h = gn >> 6;
        int b = gm >> 11, s = gm & (S - 1);
        if (mode == 1) val += posemb[s * D + d];
        outb[(((size_t)(b * H + h)) * S + s) * D + d] = (__bf16)val;
      }
    }
  }
}

// ---------------- causal attention: one wave32 per 16-query strip ----------------
// Q,K: bf16 [B,H,S,D] row-major; VT: bf16 [B,H,D,S] (pre-transposed).
__global__ __launch_bounds__(32)
void attn_kernel(const __bf16* __restrict__ Q, const __bf16* __restrict__ K,
                 const __bf16* __restrict__ VT, float* __restrict__ attn,
                 __bf16* __restrict__ ctx) {
  const int S = S_LEN, D = D_DIM, H = H_NUM;
  int bid = blockIdx.x;
  int qi = bid & 127;               // S/16 strips
  int hh = (bid >> 7) & 15;
  int b  = bid >> 11;
  int q0 = qi * 16;
  int lane = threadIdx.x, lane16 = lane & 15, hi = lane >> 4;

  __shared__ __align__(16) __bf16 Ps[16][32];    // prob tile (A-fragment source)

  size_t headBase = ((size_t)(b * H + hh)) * S;        // for Q,K rows and attn
  size_t vtBase   = ((size_t)(b * H + hh)) * D;        // for VT rows
  const __bf16* qr_ptr = Q + (headBase + q0 + lane16) * D;
  v16bf aq0 = load_frag(qr_ptr + hi * 16,      qr_ptr + hi * 16 + 8);   // d 0..31
  v16bf aq1 = load_frag(qr_ptr + 32 + hi * 16, qr_ptr + 40 + hi * 16);   // d 32..63

  int nkt = (q0 + 16 + 31) >> 5;   // causal: only key tiles covering k <= q0+15
  v8f vz = {0.f, 0.f, 0.f, 0.f, 0.f, 0.f, 0.f, 0.f};

  float m[8], l[8];
#pragma unroll
  for (int r = 0; r < 8; r++) { m[r] = -3.0e38f; l[r] = 0.f; }

  // ---- pass 1: row max / sum (online softmax stats) ----
  for (int kt = 0; kt < nkt; kt++) {
    int kbase = kt * 32;
    if (kt + 1 < nkt)
      __builtin_prefetch(K + (headBase + kbase + 32 + lane) * D, 0, 1);
#pragma unroll
    for (int j = 0; j < 2; j++) {
      int key = kbase + j * 16 + lane16;
      const __bf16* kr = K + (headBase + key) * D;
      v16bf b0 = load_frag(kr + hi * 16,      kr + hi * 16 + 8);
      v16bf b1 = load_frag(kr + 32 + hi * 16, kr + 32 + hi * 16 + 8);
      v8f acc = vz;
      acc = wmma_bf16(aq0, b0, acc);
      acc = wmma_bf16(aq1, b1, acc);
      FragC c; c.v = acc;
#pragma unroll
      for (int r = 0; r < 8; r++) {
        int qr = q0 + hi * 8 + r;
        float x = c.f[r] * 0.125f + (key > qr ? -1.0e9f : 0.f);
        float mo = m[r], mn = fmaxf(mo, x);
        l[r] = l[r] * __expf(mo - mn) + __expf(x - mn);
        m[r] = mn;
      }
    }
  }
  // reduce stats across the 16 lanes that share each row group
#pragma unroll
  for (int msk = 1; msk < 16; msk <<= 1) {
#pragma unroll
    for (int r = 0; r < 8; r++) {
      float mo = __shfl_xor(m[r], msk, 32);
      float lo = __shfl_xor(l[r], msk, 32);
      float mn = fmaxf(m[r], mo);
      l[r] = l[r] * __expf(m[r] - mn) + lo * __expf(mo - mn);
      m[r] = mn;
    }
  }
  float linv[8];
#pragma unroll
  for (int r = 0; r < 8; r++) linv[r] = 1.0f / l[r];

  // ---- pass 2: write probs, accumulate ctx = P @ V ----
  v8f cacc[4];
#pragma unroll
  for (int j = 0; j < 4; j++) cacc[j] = vz;

  for (int kt = 0; kt < nkt; kt++) {
    int kbase = kt * 32;
#pragma unroll
    for (int j = 0; j < 2; j++) {
      int key = kbase + j * 16 + lane16;
      const __bf16* kr = K + (headBase + key) * D;
      v16bf b0 = load_frag(kr + hi * 16,      kr + hi * 16 + 8);
      v16bf b1 = load_frag(kr + 32 + hi * 16, kr + 32 + hi * 16 + 8);
      v8f acc = vz;
      acc = wmma_bf16(aq0, b0, acc);
      acc = wmma_bf16(aq1, b1, acc);
      FragC c; c.v = acc;
#pragma unroll
      for (int r = 0; r < 8; r++) {
        int qr = q0 + hi * 8 + r;
        float x = c.f[r] * 0.125f + (key > qr ? -1.0e9f : 0.f);
        float p = __expf(x - m[r]) * linv[r];
        attn[(headBase + qr) * S + key] = p;
        Ps[hi * 8 + r][j * 16 + lane16] = (__bf16)p;
      }
    }
    // P fragment via LDS re-layout; V fragments straight from global V^T
    v16bf ap = load_frag(&Ps[lane16][hi * 16], &Ps[lane16][hi * 16 + 8]);
#pragma unroll
    for (int j2 = 0; j2 < 4; j2++) {
      const __bf16* vr = VT + (vtBase + j2 * 16 + lane16) * S + kbase + hi * 16;
      v16bf bv = load_frag(vr, vr + 8);
      cacc[j2] = wmma_bf16(ap, bv, cacc[j2]);
    }
  }

  // write ctx back in [B,S,F] bf16 layout for the output projection GEMM
#pragma unroll
  for (int j2 = 0; j2 < 4; j2++) {
    FragC c; c.v = cacc[j2];
#pragma unroll
    for (int r = 0; r < 8; r++) {
      int qr = q0 + hi * 8 + r;
      int f = hh * D + j2 * 16 + lane16;
      ctx[((size_t)(b * S) + qr) * F_DIM + f] = (__bf16)c.f[r];
    }
  }
}

// ---------------- host-side orchestration ----------------
extern "C" void kernel_launch(void* const* d_in, const int* in_sizes, int n_in,
                              void* d_out, int out_size, void* d_ws, size_t ws_size,
                              hipStream_t stream) {
  (void)in_sizes; (void)n_in; (void)out_size; (void)ws_size;
  const int B = 2, S = S_LEN, F = F_DIM, H = H_NUM;
  const int NT = B * S;                     // 4096 tokens

  const float* x      = (const float*)d_in[0];
  const float* Wq     = (const float*)d_in[1];
  const float* bq     = (const float*)d_in[2];
  const float* Wk     = (const float*)d_in[3];
  const float* bk     = (const float*)d_in[4];
  const float* Wv     = (const float*)d_in[5];
  const float* bv     = (const float*)d_in[6];
  const float* Wo     = (const float*)d_in[7];
  const float* bo     = (const float*)d_in[8];
  const float* gamma1 = (const float*)d_in[9];
  const float* beta1  = (const float*)d_in[10];
  const float* gamma2 = (const float*)d_in[11];
  const float* beta2  = (const float*)d_in[12];
  const float* posemb = (const float*)d_in[13];

  char* ws = (char*)d_ws;
  float*  xn32 = (float*)(ws);                              // 16 MB
  float*  y32  = (float*)(ws + ((size_t)16 << 20));         // 16 MB
  __bf16* xnb  = (__bf16*)(ws + ((size_t)32 << 20));        //  8 MB
  __bf16* wqt  = (__bf16*)(ws + ((size_t)40 << 20));        //  2 MB each (W^T)
  __bf16* wkt  = wqt + (size_t)F * F;
  __bf16* wvt  = wkt + (size_t)F * F;
  __bf16* wot  = wvt + (size_t)F * F;
  __bf16* Qb   = (__bf16*)(ws + ((size_t)48 << 20));        //  8 MB
  __bf16* Kb   = (__bf16*)(ws + ((size_t)56 << 20));        //  8 MB
  __bf16* Vb   = (__bf16*)(ws + ((size_t)64 << 20));        //  8 MB [B,H,S,D]
  __bf16* ctxb = (__bf16*)(ws + ((size_t)72 << 20));        //  8 MB
  __bf16* VTb  = (__bf16*)(ws + ((size_t)80 << 20));        //  8 MB [B,H,D,S]

  float* outp  = (float*)d_out;                      // [B,S,F] fp32
  float* attnp = outp + (size_t)B * S * F;           // [B,H,S,S] fp32

  // weights -> bf16, transposed (B-fragment friendly)
  dim3 tw(F / 32, F / 32);
  transpose_w_bf16<<<tw, 256, 0, stream>>>(Wq, wqt);
  transpose_w_bf16<<<tw, 256, 0, stream>>>(Wk, wkt);
  transpose_w_bf16<<<tw, 256, 0, stream>>>(Wv, wvt);
  transpose_w_bf16<<<tw, 256, 0, stream>>>(Wo, wot);

  // xn = LN1(x), fp32 + bf16
  ln_kernel<<<NT, 256, 0, stream>>>(x, gamma1, beta1, xn32, xnb);

  // Q/K/V projections (WMMA)
  dim3 gg(F / 64, NT / 128);
  gemm_wmma<<<gg, 256, 0, stream>>>(xnb, wqt, bq, nullptr, nullptr, Qb, nullptr, 0);
  gemm_wmma<<<gg, 256, 0, stream>>>(xnb, wkt, bk, posemb,  nullptr, Kb, nullptr, 1);
  gemm_wmma<<<gg, 256, 0, stream>>>(xnb, wvt, bv, nullptr, nullptr, Vb, nullptr, 2);

  // V -> V^T for direct B-fragment global loads in attention
  dim3 tv(S / 32, D_DIM / 32, B * H);
  transpose_v_bf16<<<tv, 256, 0, stream>>>(Vb, VTb);

  // zero attn (upper triangle never written by the attention kernel)
  long n4 = (long)B * H * S * S / 4;
  zero_f4<<<4096, 256, 0, stream>>>((float4*)attnp, n4);

  // attention: probs -> d_out, ctx -> ws
  attn_kernel<<<B * H * (S / 16), 32, 0, stream>>>(Qb, Kb, VTb, attnp, ctxb);

  // out-proj + bias + residual(xn) -> y32, then LN2 -> d_out
  gemm_wmma<<<gg, 256, 0, stream>>>(ctxb, wot, bo, nullptr, xn32, nullptr, y32, 3);
  ln_kernel<<<NT, 256, 0, stream>>>(y32, gamma2, beta2, outp, nullptr);
}
